// AMLGraphNet_72129680769136
// MI455X (gfx1250) — compile-verified, matrix operand order checked by reading, not run on hardware
//
#include <hip/hip_runtime.h>
#include <math.h>

typedef __attribute__((ext_vector_type(16))) _Float16 v16h;
typedef __attribute__((ext_vector_type(8)))  float    v8f;

#define BNS 0.9999950000374997f   // 1/sqrt(1+1e-5)

static __device__ __forceinline__ float elu1(float x) {
  return x > 0.f ? x : (__expf(x) - 1.f);
}
// order-preserving float->uint key for atomicMax-based segment max
static __device__ __forceinline__ unsigned f2key(float f) {
  unsigned u = __float_as_uint(f);
  return (u & 0x80000000u) ? ~u : (u | 0x80000000u);
}
static __device__ __forceinline__ float key2f(unsigned k) {
  unsigned u = (k & 0x80000000u) ? (k & 0x7FFFFFFFu) : ~k;
  return __uint_as_float(u);
}

// ---------------- utility kernels ----------------
__global__ void k_zero_u32(unsigned* p, long long n) {
  long long i = (long long)blockIdx.x * blockDim.x + threadIdx.x;
  if (i < n) p[i] = 0u;
}

__global__ void k_f32_to_f16(_Float16* d, const float* s, long long n) {
  long long i = (long long)blockIdx.x * blockDim.x + threadIdx.x;
  if (i < n) d[i] = (_Float16)s[i];
}

// W [K, Ncols] f32 row-major  ->  Wt [Ncols, K] f16 row-major
__global__ void k_w_to_f16_t(_Float16* dt, const float* s, int K, int Ncols) {
  int i = blockIdx.x * blockDim.x + threadIdx.x;
  if (i < K * Ncols) {
    int k = i / Ncols, n = i - k * Ncols;
    dt[(size_t)n * K + k] = (_Float16)s[i];
  }
}

// ---------------- WMMA GEMM (register-blocked: 16 x (16*NT) per wave) ----------------
// C[M, Ncols] f32 = A[M, K] f16 (row major) x Bt[Ncols, K] f16 (B transposed).
// One wave computes NT adjacent 16x16 tiles in the N dimension; the A fragment is
// loaded once per K-chunk and reused by NT back-to-back v_wmma_f32_16x16x32_f16.
// NOTE: all B accesses are derived from ONE __global base pointer with a uniform
// row-stride offset (j * 8K dwords) -- a pointer array here would decay to generic
// address space and lower to flat_load/s_wait_loadcnt_dscnt, serializing the loop.
// Requires M % 16 == 0 (N=50000=3125*16), K % 32 == 0, Ncols % (16*NT) == 0.
template <int NT>
__global__ void k_wmma_gemm(const _Float16* __restrict__ A,
                            const _Float16* __restrict__ Bt,
                            float* __restrict__ C,
                            int Mtiles, int K, int Ncols) {
  int wave = (int)(((long long)blockIdx.x * blockDim.x + threadIdx.x) >> 5);
  int lane = threadIdx.x & 31;
  int ngroups = Ncols / (16 * NT);
  int tm  = wave / ngroups;
  int tn0 = (wave - tm * ngroups) * NT;
  if (tm >= Mtiles) return;                 // wave-uniform: EXEC all-1 at WMMA
  int r  = lane & 15;
  int hs = lane >> 4;
  const unsigned* ap = (const unsigned*)(A + (size_t)(tm * 16 + r) * K);
  const unsigned* bp = (const unsigned*)(Bt + (size_t)(tn0 * 16 + r) * K);
  const size_t strideJ = (size_t)8 * K;     // 16 rows * K halfs / 2 dwords (uniform)

  v8f acc[NT];
#pragma unroll
  for (int j = 0; j < NT; ++j) acc[j] = {0.f, 0.f, 0.f, 0.f, 0.f, 0.f, 0.f, 0.f};

  for (int k0 = 0; k0 < K; k0 += 32) {
    __builtin_prefetch(ap + 16, 0, 1);      // global_prefetch_b8 next K-chunk
#pragma unroll
    for (int j = 0; j < NT; ++j) __builtin_prefetch(bp + j * strideJ + 16, 0, 1);

    union { unsigned u[8]; v16h h; } ua;
    union { unsigned u[8]; v16h h; } ub[NT];
#pragma unroll
    for (int v = 0; v < 8; ++v) {
      // dword index within 32-element K-chunk per ISA 16-bit fragment layout:
      // K = (v<4?0:16) + (hs?8:0) + 2*(v&3)  ->  dword = K/2
      int kp = ((v & 4) << 1) + (hs << 2) + (v & 3);
      ua.u[v] = ap[kp];
#pragma unroll
      for (int j = 0; j < NT; ++j) ub[j].u[v] = bp[j * strideJ + kp];
    }
    ap += 16;
    bp += 16;
#pragma unroll
    for (int j = 0; j < NT; ++j) {
      acc[j] = __builtin_amdgcn_wmma_f32_16x16x32_f16(
          false, ua.h, false, ub[j].h, (short)0, acc[j], false, false);
    }
  }
  // 32-bit C/D layout: lane -> col (r), VGPR v -> row v + 8*hs
  float* crow = C + (size_t)(tm * 16 + 8 * hs) * Ncols + r;
#pragma unroll
  for (int j = 0; j < NT; ++j)
#pragma unroll
    for (int v = 0; v < 8; ++v)
      crow[(size_t)v * Ncols + (tn0 + j) * 16] = acc[j][v];
}

// ---------------- GAT kernels ----------------
// als[n,h] = dot(z[n,h,:], avec[h,:])   (C = 64)
__global__ void k_att_dot(const float* __restrict__ z, const float* __restrict__ avec,
                          float* __restrict__ out, int N, int H) {
  int i = blockIdx.x * blockDim.x + threadIdx.x;
  if (i >= N * H) return;
  int n = i / H, h = i - n * H;
  const float* zp = z + ((size_t)n * H + h) * 64;
  const float* a  = avec + h * 64;
  float s = 0.f;
#pragma unroll
  for (int c = 0; c < 64; ++c) s += zp[c] * a[c];
  out[i] = s;
}

// per-edge leaky_relu(als[src]+ald[dst]) and segment-max into mkey[dst]
__global__ void k_edge_logits(const int* __restrict__ ei, const float* __restrict__ als,
                              const float* __restrict__ ald, float* __restrict__ evals,
                              unsigned* __restrict__ mkey, int E, int NE, int H) {
  int e = blockIdx.x * blockDim.x + threadIdx.x;
  if (e >= NE) return;
  int s = (e < E) ? ei[e]     : (e - E);   // self-loops appended
  int d = (e < E) ? ei[E + e] : (e - E);
  for (int h = 0; h < H; ++h) {
    float v = als[s * H + h] + ald[d * H + h];
    v = v > 0.f ? v : 0.2f * v;
    evals[(size_t)e * H + h] = v;
    atomicMax(&mkey[d * H + h], f2key(v));
  }
}

// ex = exp(e - max[dst]); den[dst] += ex
__global__ void k_edge_exp(const int* __restrict__ ei, float* __restrict__ evals,
                           const unsigned* __restrict__ mkey, float* __restrict__ den,
                           int E, int NE, int H) {
  int e = blockIdx.x * blockDim.x + threadIdx.x;
  if (e >= NE) return;
  int d = (e < E) ? ei[E + e] : (e - E);
  for (int h = 0; h < H; ++h) {
    float m  = key2f(mkey[d * H + h]);
    float ex = __expf(evals[(size_t)e * H + h] - m);
    evals[(size_t)e * H + h] = ex;
    atomicAdd(&den[d * H + h], ex);
  }
}

// out[dst, f] += z[src, f] * alpha(e, h=f/64)   (one thread per edge-feature)
__global__ void k_gat_msg(const int* __restrict__ ei, const float* __restrict__ evals,
                          const float* __restrict__ den, const float* __restrict__ z,
                          float* __restrict__ out, int E, int NE, int F, int H) {
  long long idx = (long long)blockIdx.x * blockDim.x + threadIdx.x;
  long long total = (long long)NE * F;
  if (idx >= total) return;
  int e = (int)(idx / F);
  int f = (int)(idx - (long long)e * F);
  int h = f >> 6;
  int s = (e < E) ? ei[e]     : (e - E);
  int d = (e < E) ? ei[E + e] : (e - E);
  float alpha = evals[(size_t)e * H + h] / (den[d * H + h] + 1e-16f);
  atomicAdd(&out[(size_t)d * F + f], z[(size_t)s * F + f] * alpha);
}

// h = elu((h + b) * (g*BNS) + be)   in place
__global__ void k_epilogue(float* __restrict__ h, const float* __restrict__ b,
                           const float* __restrict__ g, const float* __restrict__ be,
                           long long N, int F) {
  long long idx = (long long)blockIdx.x * blockDim.x + threadIdx.x;
  if (idx >= N * F) return;
  int f = (int)(idx % F);
  float x = h[idx] + b[f];
  x = x * (g[f] * BNS) + be[f];
  h[idx] = elu1(x);
}

// ---------------- GCN kernels ----------------
__global__ void k_deg(const int* __restrict__ ei, float* __restrict__ deg, int E, int NE) {
  int e = blockIdx.x * blockDim.x + threadIdx.x;
  if (e >= NE) return;
  int d = (e < E) ? ei[E + e] : (e - E);
  atomicAdd(&deg[d], 1.f);
}
__global__ void k_dinv(float* deg, int N) {
  int n = blockIdx.x * blockDim.x + threadIdx.x;
  if (n < N) deg[n] = deg[n] > 0.f ? rsqrtf(deg[n]) : 0.f;
}
__global__ void k_gcn_msg(const int* __restrict__ ei, const float* __restrict__ dinv,
                          const float* __restrict__ z, float* __restrict__ out,
                          int E, int NE, int F) {
  long long idx = (long long)blockIdx.x * blockDim.x + threadIdx.x;
  long long total = (long long)NE * F;
  if (idx >= total) return;
  int e = (int)(idx / F);
  int f = (int)(idx - (long long)e * F);
  int s = (e < E) ? ei[e]     : (e - E);
  int d = (e < E) ? ei[E + e] : (e - E);
  float norm = dinv[s] * dinv[d];
  atomicAdd(&out[(size_t)d * F + f], z[(size_t)s * F + f] * norm);
}

// ---------------- classifier ----------------
__global__ void k_classifier(const float* __restrict__ h,
    const float* __restrict__ cW1, const float* __restrict__ cb1,
    const float* __restrict__ cg1, const float* __restrict__ cbe1,
    const float* __restrict__ cW2, const float* __restrict__ cb2,
    const float* __restrict__ cg2, const float* __restrict__ cbe2,
    const float* __restrict__ cW3, const float* __restrict__ cb3,
    const float* __restrict__ cW4, const float* __restrict__ cb4,
    float* __restrict__ out, int N) {
  int n = blockIdx.x * blockDim.x + threadIdx.x;
  if (n >= N) return;
  float a[32], t1[32], t2[16], t3[8];
#pragma unroll
  for (int i = 0; i < 32; ++i) a[i] = h[(size_t)n * 32 + i];
  for (int j = 0; j < 32; ++j) {
    float s = cb1[j];
#pragma unroll
    for (int i = 0; i < 32; ++i) s += a[i] * cW1[i * 32 + j];
    s = fmaxf(s, 0.f);
    t1[j] = s * (cg1[j] * BNS) + cbe1[j];
  }
  for (int j = 0; j < 16; ++j) {
    float s = cb2[j];
#pragma unroll
    for (int i = 0; i < 32; ++i) s += t1[i] * cW2[i * 16 + j];
    s = fmaxf(s, 0.f);
    t2[j] = s * (cg2[j] * BNS) + cbe2[j];
  }
  for (int j = 0; j < 8; ++j) {
    float s = cb3[j];
#pragma unroll
    for (int i = 0; i < 16; ++i) s += t2[i] * cW3[i * 8 + j];
    t3[j] = fmaxf(s, 0.f);
  }
  float l0 = cb4[0], l1 = cb4[1];
#pragma unroll
  for (int i = 0; i < 8; ++i) { l0 += t3[i] * cW4[i * 2]; l1 += t3[i] * cW4[i * 2 + 1]; }
  float mx  = fmaxf(l0, l1);
  float lse = mx + __logf(__expf(l0 - mx) + __expf(l1 - mx));
  out[(size_t)n * 2]     = l0 - lse;
  out[(size_t)n * 2 + 1] = l1 - lse;
}

// ---------------- host orchestration ----------------
extern "C" void kernel_launch(void* const* d_in, const int* in_sizes, int n_in,
                              void* d_out, int out_size, void* d_ws, size_t ws_size,
                              hipStream_t stream) {
  (void)n_in; (void)out_size; (void)ws_size;
  const float* x    = (const float*)d_in[0];
  const int*   ei   = (const int*)d_in[1];
  const float* W1   = (const float*)d_in[2];
  const float* as1  = (const float*)d_in[3];
  const float* ad1  = (const float*)d_in[4];
  const float* b1   = (const float*)d_in[5];
  const float* g1   = (const float*)d_in[6];
  const float* be1  = (const float*)d_in[7];
  const float* W2   = (const float*)d_in[8];
  const float* as2  = (const float*)d_in[9];
  const float* ad2  = (const float*)d_in[10];
  const float* b2   = (const float*)d_in[11];
  const float* g2   = (const float*)d_in[12];
  const float* be2  = (const float*)d_in[13];
  const float* W3   = (const float*)d_in[14];
  const float* b3   = (const float*)d_in[15];
  const float* g3   = (const float*)d_in[16];
  const float* be3  = (const float*)d_in[17];
  const float* W4   = (const float*)d_in[18];
  const float* b4   = (const float*)d_in[19];
  const float* g4   = (const float*)d_in[20];
  const float* be4  = (const float*)d_in[21];
  const float* cW1  = (const float*)d_in[22];
  const float* cb1  = (const float*)d_in[23];
  const float* cg1  = (const float*)d_in[24];
  const float* cbe1 = (const float*)d_in[25];
  const float* cW2  = (const float*)d_in[26];
  const float* cb2  = (const float*)d_in[27];
  const float* cg2  = (const float*)d_in[28];
  const float* cbe2 = (const float*)d_in[29];
  const float* cW3  = (const float*)d_in[30];
  const float* cb3  = (const float*)d_in[31];
  const float* cW4  = (const float*)d_in[32];
  const float* cb4  = (const float*)d_in[33];

  const int N  = in_sizes[0] / 128;
  const int E  = in_sizes[1] / 2;
  const int NE = E + N;
  const int Mt = N / 16;          // N = 50000 = 3125 * 16

  // workspace layout (256B-aligned bump allocation; ~278 MB peak)
  char* ws = (char*)d_ws;
  size_t off = 0;
  auto alloc = [&](size_t bytes) -> char* {
    char* p = ws + off;
    off = (off + bytes + 255) & ~(size_t)255;
    return p;
  };
  float*     bufA  = (float*)alloc((size_t)N * 512 * 4);   // z (GEMM output)
  float*     bufB  = (float*)alloc((size_t)N * 512 * 4);   // aggregation/out -> h
  _Float16*  bufH  = (_Float16*)alloc((size_t)N * 512 * 2);// f16 activations
  _Float16*  bufW  = (_Float16*)alloc((size_t)512 * 512 * 2); // f16 transposed weights
  float*     als   = (float*)alloc((size_t)N * 8 * 4);
  float*     ald   = (float*)alloc((size_t)N * 8 * 4);
  unsigned*  mkey  = (unsigned*)alloc((size_t)N * 8 * 4);
  float*     den   = (float*)alloc((size_t)N * 8 * 4);
  float*     evals = (float*)alloc((size_t)NE * 8 * 4);
  float*     dinv  = (float*)alloc((size_t)N * 4);

  const int TB = 256;
  auto nb = [](long long n, int tb) { return (unsigned)((n + tb - 1) / tb); };

  // ================= GAT layer 1 (H=8, F=512, Kin=128) =================
  k_f32_to_f16<<<nb((long long)N * 128, TB), TB, 0, stream>>>(bufH, x, (long long)N * 128);
  k_w_to_f16_t<<<nb(128 * 512, TB), TB, 0, stream>>>(bufW, W1, 128, 512);
  k_wmma_gemm<4><<<nb((long long)Mt * 8 * 32, TB), TB, 0, stream>>>(bufH, bufW, bufA, Mt, 128, 512);
  k_att_dot<<<nb((long long)N * 8, TB), TB, 0, stream>>>(bufA, as1, als, N, 8);
  k_att_dot<<<nb((long long)N * 8, TB), TB, 0, stream>>>(bufA, ad1, ald, N, 8);
  k_zero_u32<<<nb((long long)N * 8, TB), TB, 0, stream>>>(mkey, (long long)N * 8);
  k_zero_u32<<<nb((long long)N * 8, TB), TB, 0, stream>>>((unsigned*)den, (long long)N * 8);
  k_zero_u32<<<nb((long long)N * 512, TB), TB, 0, stream>>>((unsigned*)bufB, (long long)N * 512);
  k_edge_logits<<<nb(NE, TB), TB, 0, stream>>>(ei, als, ald, evals, mkey, E, NE, 8);
  k_edge_exp<<<nb(NE, TB), TB, 0, stream>>>(ei, evals, mkey, den, E, NE, 8);
  k_gat_msg<<<nb((long long)NE * 512, TB), TB, 0, stream>>>(ei, evals, den, bufA, bufB, E, NE, 512, 8);
  k_epilogue<<<nb((long long)N * 512, TB), TB, 0, stream>>>(bufB, b1, g1, be1, N, 512);   // h1

  // ================= GAT layer 2 (H=4, F=256, Kin=512) =================
  k_f32_to_f16<<<nb((long long)N * 512, TB), TB, 0, stream>>>(bufH, bufB, (long long)N * 512);
  k_w_to_f16_t<<<nb(512 * 256, TB), TB, 0, stream>>>(bufW, W2, 512, 256);
  k_wmma_gemm<4><<<nb((long long)Mt * 4 * 32, TB), TB, 0, stream>>>(bufH, bufW, bufA, Mt, 512, 256);
  k_att_dot<<<nb((long long)N * 4, TB), TB, 0, stream>>>(bufA, as2, als, N, 4);
  k_att_dot<<<nb((long long)N * 4, TB), TB, 0, stream>>>(bufA, ad2, ald, N, 4);
  k_zero_u32<<<nb((long long)N * 4, TB), TB, 0, stream>>>(mkey, (long long)N * 4);
  k_zero_u32<<<nb((long long)N * 4, TB), TB, 0, stream>>>((unsigned*)den, (long long)N * 4);
  k_zero_u32<<<nb((long long)N * 256, TB), TB, 0, stream>>>((unsigned*)bufB, (long long)N * 256);
  k_edge_logits<<<nb(NE, TB), TB, 0, stream>>>(ei, als, ald, evals, mkey, E, NE, 4);
  k_edge_exp<<<nb(NE, TB), TB, 0, stream>>>(ei, evals, mkey, den, E, NE, 4);
  k_gat_msg<<<nb((long long)NE * 256, TB), TB, 0, stream>>>(ei, evals, den, bufA, bufB, E, NE, 256, 4);
  k_epilogue<<<nb((long long)N * 256, TB), TB, 0, stream>>>(bufB, b2, g2, be2, N, 256);   // h2

  // ================= GCN degree normalization (shared) =================
  k_zero_u32<<<nb(N, TB), TB, 0, stream>>>((unsigned*)dinv, N);
  k_deg<<<nb(NE, TB), TB, 0, stream>>>(ei, dinv, E, NE);
  k_dinv<<<nb(N, TB), TB, 0, stream>>>(dinv, N);

  // ================= GCN layer 3 (F=64, Kin=256) =================
  k_f32_to_f16<<<nb((long long)N * 256, TB), TB, 0, stream>>>(bufH, bufB, (long long)N * 256);
  k_w_to_f16_t<<<nb(256 * 64, TB), TB, 0, stream>>>(bufW, W3, 256, 64);
  k_wmma_gemm<4><<<nb((long long)Mt * 1 * 32, TB), TB, 0, stream>>>(bufH, bufW, bufA, Mt, 256, 64);
  k_zero_u32<<<nb((long long)N * 64, TB), TB, 0, stream>>>((unsigned*)bufB, (long long)N * 64);
  k_gcn_msg<<<nb((long long)NE * 64, TB), TB, 0, stream>>>(ei, dinv, bufA, bufB, E, NE, 64);
  k_epilogue<<<nb((long long)N * 64, TB), TB, 0, stream>>>(bufB, b3, g3, be3, N, 64);     // h3

  // ================= GCN layer 4 (F=32, Kin=64) =================
  k_f32_to_f16<<<nb((long long)N * 64, TB), TB, 0, stream>>>(bufH, bufB, (long long)N * 64);
  k_w_to_f16_t<<<nb(64 * 32, TB), TB, 0, stream>>>(bufW, W4, 64, 32);
  k_wmma_gemm<2><<<nb((long long)Mt * 1 * 32, TB), TB, 0, stream>>>(bufH, bufW, bufA, Mt, 64, 32);
  k_zero_u32<<<nb((long long)N * 32, TB), TB, 0, stream>>>((unsigned*)bufB, (long long)N * 32);
  k_gcn_msg<<<nb((long long)NE * 32, TB), TB, 0, stream>>>(ei, dinv, bufA, bufB, E, NE, 32);
  k_epilogue<<<nb((long long)N * 32, TB), TB, 0, stream>>>(bufB, b4, g4, be4, N, 32);     // h4

  // ================= classifier MLP + log_softmax =================
  k_classifier<<<nb(N, TB), TB, 0, stream>>>(bufB, cW1, cb1, cg1, cbe1,
                                             cW2, cb2, cg2, cbe2,
                                             cW3, cb3, cW4, cb4,
                                             (float*)d_out, N);
}
